// LoopStableMamba3SISO_76459007803786
// MI455X (gfx1250) — compile-verified, matrix operand order checked by reading, not run on hardware
//
#include <hip/hip_runtime.h>
#include <hip/hip_bf16.h>

// ---------------- problem constants ----------------
#define D_MODEL 1024
#define D_MAMBA 2048
#define DSTATE  16
#define N_HEADS 32
#define BATCH   4
#define SEQ     2048
#define MROWS   (BATCH * SEQ)   // 8192 token rows

typedef __bf16 bf16_t;
typedef bf16_t v16bf __attribute__((ext_vector_type(16)));
typedef float  v8f   __attribute__((ext_vector_type(8)));
typedef int    v4i   __attribute__((ext_vector_type(4)));

#if defined(__has_builtin)
#  if __has_builtin(__builtin_amdgcn_global_load_async_to_lds_b128)
#    define HAVE_ASYNC_LDS 1
#  endif
#  if __has_builtin(__builtin_amdgcn_s_wait_asynccnt)
#    define HAVE_WAIT_ASYNC_BUILTIN 1
#  endif
#endif

// ---------------------------------------------------------------------------
// fp32 -> bf16 conversion
// ---------------------------------------------------------------------------
__global__ void k_f32_to_bf16(const float* __restrict__ src,
                              bf16_t* __restrict__ dst, int n) {
    int i = blockIdx.x * blockDim.x + threadIdx.x;
    if (i < n) dst[i] = (bf16_t)src[i];
}

// ---------------------------------------------------------------------------
// WMMA fragment helpers (bf16 16x16x32).
// ISA 16x32 bf16 A layout: lane<16 holds row (lane&15), K = {0..7, 16..23};
// lane>=16 holds same row, K = {8..15, 24..31}  => two 16-byte loads.
// ---------------------------------------------------------------------------
union FragU { v16bf v; uint4 q[2]; };

__device__ inline v16bf load_frag_g(const bf16_t* __restrict__ base,
                                    int row, int K, int k0, int lane) {
    const bf16_t* p = base + (size_t)row * (size_t)K + (size_t)(k0 + ((lane >> 4) << 3));
    FragU f;
    f.q[0] = *(const uint4*)(p);
    f.q[1] = *(const uint4*)(p + 16);
    return f.v;
}

#define BS_STRIDE 40   // 32 + 8 pad elements -> 80B row stride, conflict-free

__device__ inline v16bf load_frag_lds(const bf16_t* Bs, int rowbase, int lane) {
    const bf16_t* p = Bs + (rowbase + (lane & 15)) * BS_STRIDE + ((lane >> 4) << 3);
    FragU f;
    f.q[0] = *(const uint4*)(p);
    f.q[1] = *(const uint4*)(p + 16);
    return f.v;
}

// 16-byte global -> LDS stage (async on gfx1250, sync fallback)
__device__ inline void stage16(const bf16_t* g, bf16_t* l) {
#if defined(HAVE_ASYNC_LDS)
    __builtin_amdgcn_global_load_async_to_lds_b128(
        (v4i __attribute__((address_space(1)))*)(g),
        (v4i __attribute__((address_space(3)))*)(l),
        0, 0);
#else
    *(uint4*)l = *(const uint4*)g;
#endif
}

__device__ inline void wait_async(int pending_next) {
#if defined(HAVE_WAIT_ASYNC_BUILTIN)
    if (pending_next) __builtin_amdgcn_s_wait_asynccnt(1);
    else              __builtin_amdgcn_s_wait_asynccnt(0);
#elif defined(HAVE_ASYNC_LDS)
    if (pending_next) asm volatile("s_wait_asynccnt 0x1" ::: "memory");
    else              asm volatile("s_wait_asynccnt 0x0" ::: "memory");
#else
    (void)pending_next;
#endif
}

// ---------------------------------------------------------------------------
// bf16 WMMA GEMM:  out[M,N](f32) = A[M,K](bf16) * W[N,K](bf16)^T  (+epilogue)
// block = 128 threads = 4 waves stacked in M; each wave -> 32x32 tile;
// block tile = 128x32.  Grid = (N/32, M/128); N is always a multiple of 32,
// M a multiple of 128 -> completely guard-free (EXEC stays all-ones).
// B tile (shared by all 4 waves) is double-buffered in LDS via async copy.
// mode 0: plain f32 store
// mode 1: dt epilogue: clamp(x+bias,-10,5) -> softplus -> clamp(1e-4, 0.1)
// ---------------------------------------------------------------------------
__global__ __launch_bounds__(128)
void k_gemm_wmma_bf16(const bf16_t* __restrict__ A,
                      const bf16_t* __restrict__ W,
                      float* __restrict__ out,
                      int M, int N, int K,
                      int mode, const float* __restrict__ bias) {
    __shared__ bf16_t Bs[2][32 * BS_STRIDE];   // 2 x 2.5KB

    const int lane = threadIdx.x & 31;
    const int wave = threadIdx.x >> 5;
    const int bm = blockIdx.y * 128 + wave * 32;   // wave row base
    const int bn = blockIdx.x * 32;                // block col base
    const int r  = lane & 15;

    // B-tile stager: thread t copies 16B: row tr = t/4, k-seg ts = (t%4)*8
    const int tr = threadIdx.x >> 2;
    const int ts = (threadIdx.x & 3) << 3;
    const bf16_t* wsrc = W + (size_t)(bn + tr) * (size_t)K + (size_t)ts;

    v8f acc00 = {}, acc01 = {}, acc10 = {}, acc11 = {};

    stage16(wsrc, &Bs[0][tr * BS_STRIDE + ts]);     // prefetch k0 = 0
    int buf = 0;
    for (int k0 = 0; k0 < K; k0 += 32) {
        const int more = (k0 + 32 < K);
        if (more) stage16(wsrc + k0 + 32, &Bs[buf ^ 1][tr * BS_STRIDE + ts]);
        wait_async(more);
        __syncthreads();                            // B tile visible to all waves

        v16bf a0 = load_frag_g(A, bm +      r, K, k0, lane);
        v16bf a1 = load_frag_g(A, bm + 16 + r, K, k0, lane);
        v16bf b0 = load_frag_lds(Bs[buf],  0, lane);
        v16bf b1 = load_frag_lds(Bs[buf], 16, lane);

        acc00 = __builtin_amdgcn_wmma_f32_16x16x32_bf16(false, a0, false, b0,
                                                        (short)0, acc00, false, false);
        acc10 = __builtin_amdgcn_wmma_f32_16x16x32_bf16(false, a1, false, b0,
                                                        (short)0, acc10, false, false);
        acc01 = __builtin_amdgcn_wmma_f32_16x16x32_bf16(false, a0, false, b1,
                                                        (short)0, acc01, false, false);
        acc11 = __builtin_amdgcn_wmma_f32_16x16x32_bf16(false, a1, false, b1,
                                                        (short)0, acc11, false, false);

        __syncthreads();                            // protect buf before rewrite
        buf ^= 1;
    }

    // D layout: element v of acc -> row m = v + 8*(lane>>4), col n = lane&15
    const int mlo = (lane >> 4) << 3;
    const int nlo = lane & 15;
    v8f accs[2][2] = { {acc00, acc01}, {acc10, acc11} };
    #pragma unroll
    for (int ti = 0; ti < 2; ++ti) {
        #pragma unroll
        for (int tj = 0; tj < 2; ++tj) {
            const int col = bn + tj * 16 + nlo;
            float bval = (mode == 1) ? bias[col] : 0.0f;
            #pragma unroll
            for (int v = 0; v < 8; ++v) {
                const int rowm = bm + ti * 16 + mlo + v;
                float val = accs[ti][tj][v];
                if (mode == 1) {
                    val += bval;
                    val = fminf(fmaxf(val, -10.0f), 5.0f);
                    val = __logf(1.0f + __expf(val));        // softplus
                    val = fminf(fmaxf(val, 1.0e-4f), 0.1f);
                }
                out[(size_t)rowm * (size_t)N + (size_t)col] = val;
            }
        }
    }
}

// ---------------------------------------------------------------------------
// Normalize B/C rows: bc[row, 0:16]=B, bc[row,16:32]=C; divide each 16-vector
// by max(||.||, 1).  col == lane within a wave32; xor masks <16 stay in group.
// ---------------------------------------------------------------------------
__global__ __launch_bounds__(256)
void k_bc_normalize(const float* __restrict__ bc,
                    float* __restrict__ Bm, float* __restrict__ Cm, int rows) {
    int t = blockIdx.x * blockDim.x + threadIdx.x;
    int row = t >> 5;
    int col = t & 31;
    if (row >= rows) return;
    float v = bc[(size_t)row * 32 + col];
    float ss = v * v;
    ss += __shfl_xor(ss, 1, 32);
    ss += __shfl_xor(ss, 2, 32);
    ss += __shfl_xor(ss, 4, 32);
    ss += __shfl_xor(ss, 8, 32);
    float scale = 1.0f / fmaxf(sqrtf(ss), 1.0f);
    float o = v * scale;
    if (col < 16) Bm[(size_t)row * DSTATE + col]        = o;
    else          Cm[(size_t)row * DSTATE + (col - 16)] = o;
}

// ---------------------------------------------------------------------------
// Selective scan.  One thread per (b, d, n) chain: 131072 threads, 4096 waves.
// h_t = exp(-dt*A)*h_{t-1} + (dt*x)*B ;  y = sum_n h*C + x*D  (16-lane reduce)
// Emits y directly as bf16 for the final WMMA GEMM.
// ---------------------------------------------------------------------------
__global__ __launch_bounds__(256)
void k_selective_scan(const float* __restrict__ x_ssm,
                      const float* __restrict__ dtv,
                      const float* __restrict__ Bm,
                      const float* __restrict__ Cm,
                      const float* __restrict__ A_log,
                      const float* __restrict__ Dvec,
                      bf16_t* __restrict__ y_bf16) {
    int t = blockIdx.x * blockDim.x + threadIdx.x;   // 0 .. BATCH*D_MAMBA*16-1
    int n    = t & 15;
    int chan = t >> 4;
    int b = chan / D_MAMBA;
    int d = chan % D_MAMBA;
    int head = d / (D_MAMBA / N_HEADS);

    const float Acoef = __expf(A_log[head * DSTATE + n]);
    const float Dd = Dvec[d];

    const float*  xp = x_ssm + (size_t)b * SEQ * D_MAMBA + d;
    const float*  dp = dtv   + (size_t)b * SEQ * D_MAMBA + d;
    const float*  bp = Bm    + (size_t)b * SEQ * DSTATE  + n;
    const float*  cp = Cm    + (size_t)b * SEQ * DSTATE  + n;
    bf16_t*       yp = y_bf16 + (size_t)b * SEQ * D_MAMBA + d;

    float h = 0.0f;
    for (int l = 0; l < SEQ; ++l) {
        float xv  = xp[(size_t)l * D_MAMBA];
        float dtl = dp[(size_t)l * D_MAMBA];
        float bv  = bp[(size_t)l * DSTATE];
        float cv  = cp[(size_t)l * DSTATE];
        float decay = __expf(-dtl * Acoef);
        h = fmaf(decay, h, dtl * xv * bv);
        float p = h * cv;
        p += __shfl_xor(p, 8, 32);
        p += __shfl_xor(p, 4, 32);
        p += __shfl_xor(p, 2, 32);
        p += __shfl_xor(p, 1, 32);
        if (n == 0) yp[(size_t)l * D_MAMBA] = (bf16_t)(p + xv * Dd);
    }
}

// ---------------------------------------------------------------------------
// Launcher
// Inputs: 0=x_norm 1=x_proj_w 2=dt_proj_w 3=dt_proj_b 4=B_proj_w 5=C_proj_w
//         6=A_log 7=D 8=out_proj_w   (all fp32)
// ---------------------------------------------------------------------------
extern "C" void kernel_launch(void* const* d_in, const int* in_sizes, int n_in,
                              void* d_out, int out_size, void* d_ws, size_t ws_size,
                              hipStream_t stream) {
    const float* x_norm    = (const float*)d_in[0];
    const float* x_proj_w  = (const float*)d_in[1];
    const float* dt_proj_w = (const float*)d_in[2];
    const float* dt_proj_b = (const float*)d_in[3];
    const float* B_proj_w  = (const float*)d_in[4];
    const float* C_proj_w  = (const float*)d_in[5];
    const float* A_log     = (const float*)d_in[6];
    const float* Dvec      = (const float*)d_in[7];
    const float* out_proj_w= (const float*)d_in[8];
    float* outp = (float*)d_out;

    // ---- workspace layout (256B aligned) ----
    uintptr_t p = (uintptr_t)d_ws;
    auto alloc = [&](size_t bytes) -> void* {
        void* r = (void*)p;
        p += (bytes + 255) & ~(size_t)255;
        return r;
    };
    bf16_t* xb   = (bf16_t*)alloc((size_t)MROWS * D_MODEL * 2);      // x_norm bf16
    bf16_t* wxb  = (bf16_t*)alloc((size_t)D_MAMBA * D_MODEL * 2);    // x_proj bf16
    bf16_t* wdtb = (bf16_t*)alloc((size_t)D_MAMBA * D_MODEL * 2);    // dt_proj bf16
    bf16_t* wbcb = (bf16_t*)alloc((size_t)32 * D_MODEL * 2);         // [B;C] proj bf16
    bf16_t* wob  = (bf16_t*)alloc((size_t)D_MAMBA * D_MAMBA * 2);    // out_proj bf16
    float*  xssm = (float*) alloc((size_t)MROWS * D_MAMBA * 4);      // x_ssm f32
    float*  dts  = (float*) alloc((size_t)MROWS * D_MAMBA * 4);      // dt f32
    float*  bc   = (float*) alloc((size_t)MROWS * 32 * 4);           // raw B|C f32
    float*  bm   = (float*) alloc((size_t)MROWS * DSTATE * 4);       // normalized B
    float*  cm   = (float*) alloc((size_t)MROWS * DSTATE * 4);       // normalized C
    bf16_t* yb   = (bf16_t*)alloc((size_t)MROWS * D_MAMBA * 2);      // scan output bf16

    const int CT = 256;
    auto cvt = [&](const float* s, bf16_t* d, int n) {
        k_f32_to_bf16<<<(n + CT - 1) / CT, CT, 0, stream>>>(s, d, n);
    };

    // ---- stage 0: precision conversion ----
    cvt(x_norm,    xb,   MROWS * D_MODEL);
    cvt(x_proj_w,  wxb,  D_MAMBA * D_MODEL);
    cvt(dt_proj_w, wdtb, D_MAMBA * D_MODEL);
    cvt(B_proj_w,  wbcb,                    DSTATE * D_MODEL);
    cvt(C_proj_w,  wbcb + DSTATE * D_MODEL, DSTATE * D_MODEL);
    cvt(out_proj_w, wob, D_MAMBA * D_MAMBA);

    // ---- stage 1: projections via WMMA ----
    dim3 blk(128);
    dim3 grid_main(D_MAMBA / 32, MROWS / 128);   // (64, 64)
    k_gemm_wmma_bf16<<<grid_main, blk, 0, stream>>>(
        xb, wxb, xssm, MROWS, D_MAMBA, D_MODEL, 0, nullptr);
    k_gemm_wmma_bf16<<<grid_main, blk, 0, stream>>>(
        xb, wdtb, dts, MROWS, D_MAMBA, D_MODEL, 1, dt_proj_b);
    dim3 grid_bc(1, MROWS / 128);                // N = 32 -> one col tile
    k_gemm_wmma_bf16<<<grid_bc, blk, 0, stream>>>(
        xb, wbcb, bc, MROWS, 32, D_MODEL, 0, nullptr);

    // ---- stage 2: B/C row normalization ----
    k_bc_normalize<<<(MROWS * 32) / CT, CT, 0, stream>>>(bc, bm, cm, MROWS);

    // ---- stage 3: selective scan (fp32 state, emits bf16 y) ----
    int scan_threads = BATCH * D_MAMBA * DSTATE;   // 131072
    k_selective_scan<<<scan_threads / CT, CT, 0, stream>>>(
        xssm, dts, bm, cm, A_log, Dvec, yb);

    // ---- stage 4: output projection via WMMA into d_out (f32) ----
    k_gemm_wmma_bf16<<<grid_main, blk, 0, stream>>>(
        yb, wob, outp, MROWS, D_MAMBA, D_MAMBA, 0, nullptr);

    (void)in_sizes; (void)n_in; (void)out_size; (void)ws_size;
}
